// message_28467043238252
// MI455X (gfx1250) — matrix operand order repaired and checked by reading, not compiled, not run on hardware
//
#include <hip/hip_runtime.h>
#include <hip/hip_bf16.h>
#include <math.h>

// ---------------------------------------------------------------------------
// PaiNN message layer for MI455X (gfx1250), wave32 + WMMA bf16.
//   phi = silu(s@W1+b1)@W2+b2 is per-NODE (depends only on src) -> hoisted:
//     node kernel: dense GEMMs via v_wmma_f32_16x16x32_bf16, P = phi  (N x 384)
//   edge kernel: W = (rbf@Wr + br)*fcut via WMMA (K padded 20->32),
//     split = W * P[src], scatter-add with global_atomic_add_f32.
// ---------------------------------------------------------------------------

#define RCUT  5.0f
#define PI_F  3.14159265358979323846f
#define FDIM  128
#define F3    384

typedef __attribute__((ext_vector_type(16))) __bf16 v16bf;
typedef __attribute__((ext_vector_type(8)))  float  v8f;

union BF16x16 { v16bf v; unsigned int u[8]; uint4 q[2]; };

// One-instruction packed convert, pinned ordering (lo -> [15:0], hi -> [31:16]).
// Inline asm keeps the SLP vectorizer from re-pairing + v_perm_b32 fixups.
__device__ __forceinline__ unsigned int pack2bf(float lo, float hi) {
  unsigned int r;
  asm("v_cvt_pk_bf16_f32 %0, %1, %2" : "=v"(r) : "v"(lo), "v"(hi));
  return r;
}

__device__ __forceinline__ unsigned short f2bf(float f) {
  __bf16 b = (__bf16)f;
  return __builtin_bit_cast(unsigned short, b);
}

// ---------------------------------------------------------------------------
__global__ void zero_f4(float4* __restrict__ p, long n4) {
  long i = (long)blockIdx.x * blockDim.x + threadIdx.x;
  long stride = (long)gridDim.x * blockDim.x;
  float4 z = make_float4(0.f, 0.f, 0.f, 0.f);
  for (; i < n4; i += stride) p[i] = z;
}

// ---------------------------------------------------------------------------
// Pack W1(128x128), W2(128x384), Wr(20x384 padded K->32) into bf16 B-operand
// tiles. Tile = 512 bf16: lane (0..31) x j (0..15). B layout (16-bit, 32x16):
//   col = lane&15 ; K = (lane>>4)*8 + j + (j>=8)*8
__global__ void pack_weights(const float* __restrict__ W1,
                             const float* __restrict__ W2,
                             const float* __restrict__ Wr,
                             unsigned short* __restrict__ W1p,
                             unsigned short* __restrict__ W2p,
                             unsigned short* __restrict__ Wrp) {
  const int idx = blockIdx.x * blockDim.x + threadIdx.x;
  const int t1 = 32 * 512, t2 = 96 * 512, t3 = 24 * 512;
  if (idx < t1) {                       // W1: kc(0..3) x nt(0..7)
    int t = idx >> 9, rem = idx & 511;
    int lane = rem >> 4, j = rem & 15;
    int kc = t >> 3, nt = t & 7;
    int col = nt * 16 + (lane & 15);
    int k = kc * 32 + (lane >> 4) * 8 + j + (j >= 8 ? 8 : 0);
    W1p[idx] = f2bf(W1[k * FDIM + col]);
  } else if (idx < t1 + t2) {           // W2: kc(0..3) x nt(0..23), t = kc*24+nt
    int id = idx - t1;
    int t = id >> 9, rem = id & 511;
    int lane = rem >> 4, j = rem & 15;
    int kc = t / 24, nt = t % 24;
    int col = nt * 16 + (lane & 15);
    int k = kc * 32 + (lane >> 4) * 8 + j + (j >= 8 ? 8 : 0);
    W2p[id] = f2bf(W2[k * F3 + col]);
  } else if (idx < t1 + t2 + t3) {      // Wr: nt(0..23), K padded to 32
    int id = idx - t1 - t2;
    int t = id >> 9, rem = id & 511;
    int lane = rem >> 4, j = rem & 15;
    int col = t * 16 + (lane & 15);
    int k = (lane >> 4) * 8 + j + (j >= 8 ? 8 : 0);
    Wrp[id] = (k < 20) ? f2bf(Wr[k * F3 + col]) : (unsigned short)0;
  }
}

// ---------------------------------------------------------------------------
// Node kernel: P[n,:] = silu(s[n,:]@W1+b1)@W2+b2.  One wave = 16 nodes.
__global__ __launch_bounds__(256) void painn_node_phi(
    const float* __restrict__ s, const float* __restrict__ b1,
    const float* __restrict__ b2,
    const unsigned short* __restrict__ W1p,
    const unsigned short* __restrict__ W2p,
    float* __restrict__ P, int N) {
  __shared__ float Hsh[8 * 16 * 132];   // per-wave 16x128 f32, stride 132 (=4 mod 64)
  const int lane = threadIdx.x & 31;
  const int wave = threadIdx.x >> 5;
  const int row_l = lane & 15;
  const int hf = lane >> 4;
  const int kb = hf * 8;                // A-layout: lanes 0-15 K {0..7,16..23}; 16-31 +8
  const int tile = blockIdx.x * 8 + wave;
  const int node = tile * 16 + row_l;
  const int node_c = node < N ? node : (N - 1);
  const float* srow = s + (long)node_c * FDIM;

  BF16x16 A[4];
#pragma unroll
  for (int kc = 0; kc < 4; ++kc) {
    const int b0 = kc * 32 + kb;
    float4 f0 = *(const float4*)(srow + b0);
    float4 f1 = *(const float4*)(srow + b0 + 4);
    float4 g0 = *(const float4*)(srow + b0 + 16);
    float4 g1 = *(const float4*)(srow + b0 + 20);
    A[kc].u[0] = pack2bf(f0.x, f0.y); A[kc].u[1] = pack2bf(f0.z, f0.w);
    A[kc].u[2] = pack2bf(f1.x, f1.y); A[kc].u[3] = pack2bf(f1.z, f1.w);
    A[kc].u[4] = pack2bf(g0.x, g0.y); A[kc].u[5] = pack2bf(g0.z, g0.w);
    A[kc].u[6] = pack2bf(g1.x, g1.y); A[kc].u[7] = pack2bf(g1.z, g1.w);
  }

  float* H = Hsh + wave * (16 * 132);

  // GEMM1 + bias + SiLU -> LDS (f32)
#pragma unroll
  for (int nt = 0; nt < 8; ++nt) {
    v8f acc = {};
#pragma unroll
    for (int kc = 0; kc < 4; ++kc) {
      BF16x16 B;
      const uint4* bp = (const uint4*)(W1p + (long)(kc * 8 + nt) * 512 + lane * 16);
      B.q[0] = bp[0]; B.q[1] = bp[1];
      acc = __builtin_amdgcn_wmma_f32_16x16x32_bf16(false, A[kc].v, false, B.v,
                                                    (short)0, acc, false, false);
    }
    const int col = nt * 16 + row_l;
    const float bias = b1[col];
#pragma unroll
    for (int i = 0; i < 8; ++i) {       // C layout: row = hf*8+i, col = lane&15 (+nt*16)
      float x = acc[i] + bias;
      H[(hf * 8 + i) * 132 + col] = x / (1.0f + __expf(-x));
    }
  }
  __syncthreads();

  // Re-read H in A-operand layout (cross-lane transpose via LDS)
  BF16x16 A2[4];
  const float* hrow = H + row_l * 132;
#pragma unroll
  for (int kc = 0; kc < 4; ++kc) {
    const int b0 = kc * 32 + kb;
    float4 f0 = *(const float4*)(hrow + b0);
    float4 f1 = *(const float4*)(hrow + b0 + 4);
    float4 g0 = *(const float4*)(hrow + b0 + 16);
    float4 g1 = *(const float4*)(hrow + b0 + 20);
    A2[kc].u[0] = pack2bf(f0.x, f0.y); A2[kc].u[1] = pack2bf(f0.z, f0.w);
    A2[kc].u[2] = pack2bf(f1.x, f1.y); A2[kc].u[3] = pack2bf(f1.z, f1.w);
    A2[kc].u[4] = pack2bf(g0.x, g0.y); A2[kc].u[5] = pack2bf(g0.z, g0.w);
    A2[kc].u[6] = pack2bf(g1.x, g1.y); A2[kc].u[7] = pack2bf(g1.z, g1.w);
  }

  // GEMM2 -> P (N x 384); unroll 2 => two independent WMMA accumulation chains
#pragma unroll 2
  for (int nt = 0; nt < 24; ++nt) {
    v8f acc = {};
#pragma unroll
    for (int kc = 0; kc < 4; ++kc) {
      BF16x16 B;
      const uint4* bp = (const uint4*)(W2p + (long)(kc * 24 + nt) * 512 + lane * 16);
      B.q[0] = bp[0]; B.q[1] = bp[1];
      acc = __builtin_amdgcn_wmma_f32_16x16x32_bf16(false, A2[kc].v, false, B.v,
                                                    (short)0, acc, false, false);
    }
    const int col = nt * 16 + row_l;
    const float bias = b2[col];
#pragma unroll
    for (int i = 0; i < 8; ++i) {
      int r = tile * 16 + hf * 8 + i;
      if (r < N) P[(long)r * F3 + col] = acc[i] + bias;
    }
  }
}

// ---------------------------------------------------------------------------
// Edge kernel: one wave = 16 edges. rbf filter via WMMA, fused gather/scatter.
__global__ __launch_bounds__(256) void painn_edge(
    const float* __restrict__ vfeat, const float* __restrict__ rij,
    const int* __restrict__ src, const int* __restrict__ dst,
    const float* __restrict__ br, const unsigned short* __restrict__ Wrp,
    const float* __restrict__ P,
    float* __restrict__ out_v, float* __restrict__ out_s, int E) {
  __shared__ float sh_fc[8][16], sh_dx[8][16], sh_dy[8][16], sh_dz[8][16];
  __shared__ int sh_sr[8][16], sh_ds[8][16];

  const int lane = threadIdx.x & 31;
  const int wave = threadIdx.x >> 5;
  const int row_l = lane & 15;
  const int hf = lane >> 4;
  const int tile = blockIdx.x * 8 + wave;
  const int e = tile * 16 + row_l;
  const int ec = e < E ? e : (E - 1);

  float rx = rij[(long)ec * 3 + 0];
  float ry = rij[(long)ec * 3 + 1];
  float rz = rij[(long)ec * 3 + 2];
  float d = sqrtf(rx * rx + ry * ry + rz * rz);
  float dinv = 1.0f / fmaxf(d, 1e-20f);
  // fcut==0 kills every contribution of padded/out-of-cutoff edges.
  float fc = (d < RCUT && e < E) ? 0.5f * (__cosf((PI_F / RCUT) * d) + 1.0f) : 0.0f;

  if (hf == 0) {
    sh_fc[wave][row_l] = fc;
    sh_dx[wave][row_l] = rx * dinv;
    sh_dy[wave][row_l] = ry * dinv;
    sh_dz[wave][row_l] = rz * dinv;
    sh_sr[wave][row_l] = src[ec];
    sh_ds[wave][row_l] = dst[ec];
  }

  // rbf A-operand (16x32 bf16): K>=20 zero padded.  K(j) = hf*8 + j + (j>=8)*8
  BF16x16 rbfA;
#pragma unroll
  for (int jj = 0; jj < 8; ++jj) {
    int j0 = 2 * jj, j1 = 2 * jj + 1;
    int k0 = hf * 8 + j0 + (j0 >= 8 ? 8 : 0);
    int k1 = hf * 8 + j1 + (j1 >= 8 ? 8 : 0);
    float r0 = (k0 < 20) ? __sinf((float)(k0 + 1) * (PI_F / RCUT) * d) * dinv : 0.0f;
    float r1 = (k1 < 20) ? __sinf((float)(k1 + 1) * (PI_F / RCUT) * d) * dinv : 0.0f;
    rbfA.u[jj] = pack2bf(r0, r1);
  }
  __syncthreads();

  float fc8[8], dx8[8], dy8[8], dz8[8]; int sr8[8], ds8[8];
#pragma unroll
  for (int i = 0; i < 8; ++i) {
    int r = hf * 8 + i;
    fc8[i] = sh_fc[wave][r]; dx8[i] = sh_dx[wave][r];
    dy8[i] = sh_dy[wave][r]; dz8[i] = sh_dz[wave][r];
    sr8[i] = sh_sr[wave][r]; ds8[i] = sh_ds[wave][r];
  }

  // a (cols 0..127) and c (cols 256..383) -> out_v
  for (int nt = 0; nt < 8; ++nt) {
    BF16x16 Ba, Bc;
    { const uint4* bp = (const uint4*)(Wrp + (long)nt * 512 + lane * 16);
      Ba.q[0] = bp[0]; Ba.q[1] = bp[1]; }
    { const uint4* bp = (const uint4*)(Wrp + (long)(nt + 16) * 512 + lane * 16);
      Bc.q[0] = bp[0]; Bc.q[1] = bp[1]; }
    v8f Wa = {}, Wc = {};
    Wa = __builtin_amdgcn_wmma_f32_16x16x32_bf16(false, rbfA.v, false, Ba.v,
                                                 (short)0, Wa, false, false);
    Wc = __builtin_amdgcn_wmma_f32_16x16x32_bf16(false, rbfA.v, false, Bc.v,
                                                 (short)0, Wc, false, false);
    const int colA = nt * 16 + row_l;
    const int colC = colA + 256;
    const float brA = br[colA];
    const float brC = br[colC];
#pragma unroll
    for (int i = 0; i < 8; ++i) {
      float av = (Wa[i] + brA) * fc8[i];
      float cv = (Wc[i] + brC) * fc8[i];
      const float* Pr = P + (long)sr8[i] * F3;
      float a_ = av * Pr[colA];
      float c_ = cv * Pr[colC];
      const float* vr = vfeat + (long)sr8[i] * F3;
      float* ov = out_v + (long)ds8[i] * F3;
      float m0 = a_ * vr[colA]       + c_ * dx8[i];
      float m1 = a_ * vr[128 + colA] + c_ * dy8[i];
      float m2 = a_ * vr[256 + colA] + c_ * dz8[i];
      __hip_atomic_fetch_add(ov + colA,       m0, __ATOMIC_RELAXED, __HIP_MEMORY_SCOPE_AGENT);
      __hip_atomic_fetch_add(ov + 128 + colA, m1, __ATOMIC_RELAXED, __HIP_MEMORY_SCOPE_AGENT);
      __hip_atomic_fetch_add(ov + 256 + colA, m2, __ATOMIC_RELAXED, __HIP_MEMORY_SCOPE_AGENT);
    }
  }

  // b (cols 128..255) -> out_s; unroll 2 for independent WMMA chains
#pragma unroll 2
  for (int nt = 8; nt < 16; ++nt) {
    BF16x16 Bb;
    { const uint4* bp = (const uint4*)(Wrp + (long)nt * 512 + lane * 16);
      Bb.q[0] = bp[0]; Bb.q[1] = bp[1]; }
    v8f Wb = {};
    Wb = __builtin_amdgcn_wmma_f32_16x16x32_bf16(false, rbfA.v, false, Bb.v,
                                                 (short)0, Wb, false, false);
    const int col = nt * 16 + row_l;
    const float brB = br[col];
#pragma unroll
    for (int i = 0; i < 8; ++i) {
      float b_ = (Wb[i] + brB) * fc8[i] * P[(long)sr8[i] * F3 + col];
      __hip_atomic_fetch_add(out_s + (long)ds8[i] * FDIM + (col - 128), b_,
                             __ATOMIC_RELAXED, __HIP_MEMORY_SCOPE_AGENT);
    }
  }
}

// ---------------------------------------------------------------------------
extern "C" void kernel_launch(void* const* d_in, const int* in_sizes, int n_in,
                              void* d_out, int out_size, void* d_ws, size_t ws_size,
                              hipStream_t stream) {
  const float* s   = (const float*)d_in[0];
  const float* v   = (const float*)d_in[1];
  const float* rij = (const float*)d_in[2];
  const int*   src = (const int*)d_in[3];
  const int*   dst = (const int*)d_in[4];
  const float* W1  = (const float*)d_in[5];
  const float* b1  = (const float*)d_in[6];
  const float* W2  = (const float*)d_in[7];
  const float* b2  = (const float*)d_in[8];
  const float* Wr  = (const float*)d_in[9];
  const float* br  = (const float*)d_in[10];
  const int N = in_sizes[0] / FDIM;   // n_nodes (scalar is device-resident; derive)
  const int E = in_sizes[3];

  float* out_v = (float*)d_out;                 // N*3*F
  float* out_s = out_v + (long)N * F3;          // N*F

  // ws: [W1p 32KiB bf16][W2p 96KiB][Wrp 24KiB][P = N*384 f32 (~76.8 MB)]
  unsigned short* W1p = (unsigned short*)d_ws;
  unsigned short* W2p = W1p + 32 * 512;
  unsigned short* Wrp = W2p + 96 * 512;
  float* P = (float*)((char*)d_ws + (long)(32 + 96 + 24) * 512 * sizeof(unsigned short));

  long n4 = ((long)N * 512) / 4;
  zero_f4<<<4096, 256, 0, stream>>>((float4*)d_out, n4);
  pack_weights<<<(152 * 512 + 255) / 256, 256, 0, stream>>>(W1, W2, Wr, W1p, W2p, Wrp);
  painn_node_phi<<<(N + 127) / 128, 256, 0, stream>>>(s, b1, b2, W1p, W2p, P, N);
  painn_edge<<<(E + 127) / 128, 256, 0, stream>>>(v, rij, src, dst, br, Wrp, P,
                                                  out_v, out_s, E);
  (void)n_in; (void)out_size; (void)ws_size;
}
